// BaseEncoder_6116033429939
// MI455X (gfx1250) — matrix operand order repaired and checked by reading, not compiled, not run on hardware
//
#include <hip/hip_runtime.h>
#include <hip/hip_bf16.h>
#include <math.h>
#include <stdint.h>

// Problem dims (fixed by the reference)
#define B_  2
#define Q_  4
#define L_  1024
#define H_  768
#define E_  32
#define M_  2
#define R_  5
#define P_  10
#define NH_ 12
#define BQ_ (B_ * Q_)
#define EE_ (E_ * E_)     // 1024 candidate pairs
#define H2_ (2 * H_)      // 1536

// LDS B-panel staging: 256 K-rows x 16 N-cols f32 per buffer, padded to
// 17 dwords/row (TDM pad: +1 dword per 16 -> half-wave ds reads hit
// disjoint bank sets). Two buffers for TDM double-buffering: 34 KB total.
#define KSTAGE_   256
#define BPITCH_   17

typedef __bf16 bf16_t;
typedef __attribute__((ext_vector_type(16))) __bf16 v16bf;
typedef __attribute__((ext_vector_type(8)))  float  v8f;
typedef __attribute__((ext_vector_type(4)))  unsigned int u32x4;
typedef __attribute__((ext_vector_type(8)))  int i32x8;
typedef __attribute__((ext_vector_type(4)))  int i32x4;

// ---------------------------------------------------------------------------
// WMMA fragment helpers (ISA 7.12.2, wave32):
//  A 16x32 bf16: lanes 0-15 -> M=lane, kb=0 ; lanes 16-31 -> M=lane-16, kb=8
//                element i<8 -> K=kb+i ; element 8+i -> K=16+kb+i
//  B 32x16 bf16: lanes 0-15 -> N=lane, kb=0 ; lanes 16-31 -> N=lane-16, kb=8 (K as for A)
//  C/D 16x16 f32: VGPR r: lanes 0-15 -> M=r, N=lane ; lanes 16-31 -> M=8+r, N=lane-16
// ---------------------------------------------------------------------------

__device__ __forceinline__ v16bf load_a_row(const bf16_t* __restrict__ rowp) {
  // rowp = &A[row][k0]; apply per-lane kb offset
  const int lane = threadIdx.x & 31;
  const int kb   = (lane & 16) ? 8 : 0;
  const bf16_t* p = rowp + kb;
  v16bf a;
#pragma unroll
  for (int i = 0; i < 8; ++i) { a[i] = p[i]; a[8 + i] = p[16 + i]; }
  return a;
}

__device__ __forceinline__ v16bf load_a_bf(const bf16_t* __restrict__ A, int lda,
                                           int row0, int k0) {
  const int lane = threadIdx.x & 31;
  return load_a_row(A + (size_t)(row0 + (lane & 15)) * lda + k0);
}

// B fragment from the padded LDS panel (f32 -> bf16 convert in VALU)
__device__ __forceinline__ v16bf load_b_lds(const float* __restrict__ Bl, int k_local0) {
  const int lane  = threadIdx.x & 31;
  const int n     = lane & 15;
  const int kbase = (lane & 16) ? 8 : 0;
  const float* p = Bl + (size_t)(k_local0 + kbase) * BPITCH_ + n;
  v16bf b;
#pragma unroll
  for (int i = 0; i < 16; ++i) b[i] = (bf16_t)p[(i < 8 ? i : i + 8) * BPITCH_];
  return b;
}

// ---------------------------------------------------------------------------
// TDM: DMA a [rows x 16] f32 tile (row stride = stride_elems) from global
// into LDS at lds_off, padding each 16-dword row with 1 dword (pitch 17).
// D# layout per ISA 8.3/8.4 (group0 128b, group1 256b, groups2/3 zero = 2D).
// ---------------------------------------------------------------------------
__device__ __forceinline__ void tdm_load_panel(unsigned lds_off, const void* gptr,
                                               int rows, int stride_elems) {
  const unsigned long long ga = (unsigned long long)(size_t)gptr;
  u32x4 g0;
  g0[0] = 1u;                                        // count=1, user descriptor
  g0[1] = lds_off;                                   // lds_addr (bytes)
  g0[2] = (unsigned)ga;                              // global_addr[31:0]
  g0[3] = (unsigned)((ga >> 32) & 0x1FFFFFFu)        // global_addr[56:32]
          | (2u << 30);                              // type = 2 ("image")
  i32x8 g1;
  g1[0] = (int)((2u << 16)                           // data_size = 4B
          | (1u << 20)                               // pad_enable
          | (3u << 22)                               // pad_interval: 16 dwords
          | (0u << 25));                             // pad_amount: 1 dword
  g1[1] = (int)(((unsigned)stride_elems & 0xFFFFu) << 16);      // tensor_dim0[15:0]
  g1[2] = (int)((((unsigned)stride_elems >> 16) & 0xFFFFu)      // tensor_dim0[31:16]
          | ((unsigned)rows << 16));                            // tensor_dim1[15:0]
  g1[3] = (int)(16u << 16);                          // tile_dim0 = 16
  g1[4] = rows;                                      // tile_dim1 = rows, tile_dim2 = 0
  g1[5] = stride_elems;                              // tensor_dim0_stride[31:0]
  g1[6] = 0;                                         // stride hi / dim1_stride lo
  g1[7] = 0;
  i32x4 g2 = {0, 0, 0, 0};
  i32x4 g3 = {0, 0, 0, 0};
#if __clang_major__ >= 23
  i32x8 g4 = {0, 0, 0, 0, 0, 0, 0, 0};
  __builtin_amdgcn_tensor_load_to_lds(g0, g1, g2, g3, g4, 0);
#else
  __builtin_amdgcn_tensor_load_to_lds(g0, g1, g2, g3, 0);
#endif
}

// ---------------------------------------------------------------------------
// k1: entity means (bf16) + gathered mean attention rows (f32)
// ---------------------------------------------------------------------------
__global__ void k1_gather(const float* __restrict__ seq, const float* __restrict__ att,
                          const int* __restrict__ pos,
                          bf16_t* __restrict__ ent_bf, float* __restrict__ ea) {
  const int bq = blockIdx.x;                                  // 0..7
  const float* seqq = seq + (size_t)bq * L_ * H_;
  const float* attq = att + (size_t)bq * NH_ * L_ * L_;
  const int*   posq = pos + bq * E_ * M_;
  bf16_t* entq = ent_bf + (size_t)bq * E_ * H_;
  float*  eaq  = ea + (size_t)bq * E_ * NH_ * L_;

  for (int idx = threadIdx.x; idx < E_ * H_; idx += blockDim.x) {
    const int e = idx / H_, h = idx % H_;
    float s = 0.f;
#pragma unroll
    for (int m = 0; m < M_; ++m) s += seqq[(size_t)posq[e * M_ + m] * H_ + h];
    entq[idx] = (bf16_t)(s * (1.0f / M_));
  }
  for (int idx = threadIdx.x; idx < E_ * NH_ * L_; idx += blockDim.x) {
    const int e  = idx / (NH_ * L_);
    const int hh = (idx / L_) % NH_;
    const int l  = idx % L_;
    float s = 0.f;
#pragma unroll
    for (int m = 0; m < M_; ++m)
      s += attq[((size_t)hh * L_ + posq[e * M_ + m]) * L_ + l];
    eaq[idx] = s * (1.0f / M_);
  }
}

// ---------------------------------------------------------------------------
// k2: mul[ef,l] = (1/NH) * sum_h ea[e,h,l]*ea[f,h,l]  (bf16) + row sums S[ef]
// ---------------------------------------------------------------------------
__global__ void k2_pairatt(const float* __restrict__ ea,
                           bf16_t* __restrict__ mul_bf, float* __restrict__ S) {
  const int gid = blockIdx.x;               // bq*1024 + ef
  const int bq  = gid >> 10;
  const int ef  = gid & 1023;
  const int e = ef >> 5, f = ef & 31;
  const float* ea_e = ea + (size_t)(bq * E_ + e) * NH_ * L_;
  const float* ea_f = ea + (size_t)(bq * E_ + f) * NH_ * L_;
  bf16_t* out = mul_bf + (size_t)gid * L_;

  __shared__ float red[256];
  float local = 0.f;
  for (int l = threadIdx.x; l < L_; l += blockDim.x) {
    float s = 0.f;
#pragma unroll
    for (int hh = 0; hh < NH_; ++hh)
      s += ea_e[hh * L_ + l] * ea_f[hh * L_ + l];
    s *= (1.0f / NH_);
    out[l] = (bf16_t)s;
    local += s;
  }
  red[threadIdx.x] = local;
  __syncthreads();
  for (int off = 128; off > 0; off >>= 1) {
    if ((int)threadIdx.x < off) red[threadIdx.x] += red[threadIdx.x + off];
    __syncthreads();
  }
  if (threadIdx.x == 0) S[gid] = red[0];
}

// ---------------------------------------------------------------------------
// k3: ctx[ef,h] = (1/S[ef]) * sum_l mul[ef,l] * seq[l,h]     (bf16 out)
//     Block = 128 M-rows x 16 N-cols; 8 waves, one 16x16 tile each.
//     B panel (seq[:, col0:col0+16]) staged via double-buffered TDM:
//     DMA for stage s+1 is in flight while waves run WMMA on stage s.
// ---------------------------------------------------------------------------
__global__ void k3_ctx(const bf16_t* __restrict__ mul_bf, const float* __restrict__ seq,
                       const float* __restrict__ S, bf16_t* __restrict__ ctx_bf) {
  __shared__ float Bsh[2][KSTAGE_ * BPITCH_];         // 2 x 17 KB

  const int blk = blockIdx.x;
  const int bq  = blk / (8 * 48);                     // 8 M-groups * 48 N-tiles
  const int t   = blk % (8 * 48);
  const int mg  = t / 48, nt = t % 48;
  const int wave = threadIdx.x >> 5;
  const int row0 = mg * 128 + wave * 16;
  const int col0 = nt * 16;

  const bf16_t* A  = mul_bf + (size_t)bq * EE_ * L_;
  const float*  Bm = seq    + (size_t)bq * L_ * H_;

  constexpr int NSTAGE = L_ / KSTAGE_;                // 4
  if (wave == 0)
    tdm_load_panel((unsigned)(size_t)&Bsh[0][0], Bm + col0, KSTAGE_, H_);

  v8f acc = {};
  for (int stage = 0; stage < NSTAGE; ++stage) {
    const int buf = stage & 1;
    if (wave == 0) {
      if (stage + 1 < NSTAGE) { // kick DMA for next stage into other buffer
        tdm_load_panel((unsigned)(size_t)&Bsh[buf ^ 1][0],
                       Bm + (size_t)(stage + 1) * KSTAGE_ * H_ + col0, KSTAGE_, H_);
        // TDM ops of one wave complete in order: cnt<=1 => stage's DMA done
        __builtin_amdgcn_s_wait_tensorcnt(1);
      } else {
        __builtin_amdgcn_s_wait_tensorcnt(0);
      }
    }
    __syncthreads();            // panel `stage` visible to all waves
#pragma unroll 4
    for (int kk = 0; kk < KSTAGE_; kk += 32) {
      if (kk + 32 < KSTAGE_)    // near-scope prefetch of next A chunk
        __builtin_prefetch(A + (size_t)(row0 + (threadIdx.x & 15)) * L_ +
                               stage * KSTAGE_ + kk + 32, 0, 3);
      v16bf a = load_a_bf(A, L_, row0, stage * KSTAGE_ + kk);
      v16bf b = load_b_lds(&Bsh[buf][0], kk);
      acc = __builtin_amdgcn_wmma_f32_16x16x32_bf16(false, a, false, b,
                                                    (short)0, acc, false, false);
    }
    __syncthreads();            // all reads of this buffer done before reuse
  }

  const int lane = threadIdx.x & 31;
  const int n    = lane & 15;
  const int moff = (lane & 16) ? 8 : 0;
  const float* Sq = S + bq * EE_;
  bf16_t* C = ctx_bf + (size_t)bq * EE_ * H_;
#pragma unroll
  for (int r = 0; r < 8; ++r) {
    const int m = row0 + moff + r;
    C[(size_t)m * H_ + col0 + n] = (bf16_t)(acc[r] / Sq[m]);
  }
}

// ---------------------------------------------------------------------------
// k4: cand[ef, which*H + h] = tanh( [ent_row | ctx_row] @ W )   (bf16 out)
//     Same blocking as k3; K=1536 in six 256-row double-buffered TDM stages.
//     The concat boundary (H=768) is 32-aligned: each K-step reads one source.
// ---------------------------------------------------------------------------
__global__ void k4_proj(const bf16_t* __restrict__ ent_bf, const bf16_t* __restrict__ ctx_bf,
                        const float* __restrict__ W_head, const float* __restrict__ W_tail,
                        bf16_t* __restrict__ cand_bf) {
  __shared__ float Bsh[2][KSTAGE_ * BPITCH_];         // 2 x 17 KB

  const int blk   = blockIdx.x;
  const int which = blk / (BQ_ * 8 * 48);             // 0=head, 1=tail
  const int rem   = blk % (BQ_ * 8 * 48);
  const int bq = rem / (8 * 48);
  const int t  = rem % (8 * 48);
  const int mg = t / 48, nt = t % 48;
  const int wave = threadIdx.x >> 5;
  const int row0 = mg * 128 + wave * 16;
  const int col0 = nt * 16;

  const bf16_t* entq = ent_bf + (size_t)bq * E_ * H_;
  const bf16_t* ctxq = ctx_bf + (size_t)bq * EE_ * H_;
  const float*  W    = which ? W_tail : W_head;

  const int lane = threadIdx.x & 31;
  const int ef   = row0 + (lane & 15);
  const int er   = which ? (ef & 31) : (ef >> 5);
  const bf16_t* entRow = entq + (size_t)er * H_;
  const bf16_t* ctxRow = ctxq + (size_t)ef * H_;

  constexpr int NSTAGE = H2_ / KSTAGE_;               // 6
  if (wave == 0)
    tdm_load_panel((unsigned)(size_t)&Bsh[0][0], W + col0, KSTAGE_, H_);

  v8f acc = {};
  for (int stage = 0; stage < NSTAGE; ++stage) {
    const int buf = stage & 1;
    if (wave == 0) {
      if (stage + 1 < NSTAGE) {
        tdm_load_panel((unsigned)(size_t)&Bsh[buf ^ 1][0],
                       W + (size_t)(stage + 1) * KSTAGE_ * H_ + col0, KSTAGE_, H_);
        __builtin_amdgcn_s_wait_tensorcnt(1);
      } else {
        __builtin_amdgcn_s_wait_tensorcnt(0);
      }
    }
    __syncthreads();
#pragma unroll 4
    for (int kk = 0; kk < KSTAGE_; kk += 32) {
      const int k0 = stage * KSTAGE_ + kk;
      const bf16_t* arow = (k0 < H_) ? (entRow + k0) : (ctxRow + (k0 - H_));
      v16bf a = load_a_row(arow);
      v16bf b = load_b_lds(&Bsh[buf][0], kk);
      acc = __builtin_amdgcn_wmma_f32_16x16x32_bf16(false, a, false, b,
                                                    (short)0, acc, false, false);
    }
    __syncthreads();
  }

  const int n    = lane & 15;
  const int moff = (lane & 16) ? 8 : 0;
  bf16_t* out = cand_bf + (size_t)bq * EE_ * H2_;
#pragma unroll
  for (int r = 0; r < 8; ++r) {
    const int m = row0 + moff + r;
    out[(size_t)m * H2_ + which * H_ + col0 + n] = (bf16_t)tanhf(acc[r]);
  }
}

// ---------------------------------------------------------------------------
// k5: scores[bq,n,r] = max_p <proto[b,r,p,:], cand[bq,n,:]>   (one wave per n)
// ---------------------------------------------------------------------------
__global__ void k5_score(const bf16_t* __restrict__ cand_bf, const float* __restrict__ proto,
                         float* __restrict__ out) {
  const int w  = blockIdx.x * (blockDim.x >> 5) + (threadIdx.x >> 5);
  const int bq = w >> 10;
  const int n  = w & 1023;
  const int b  = bq / Q_;
  const int lane = threadIdx.x & 31;

  const bf16_t* cr = cand_bf + ((size_t)bq * EE_ + n) * H2_;
  float cv[H2_ / 32];
#pragma unroll
  for (int i = 0; i < H2_ / 32; ++i) cv[i] = (float)cr[lane + 32 * i];

  const float* pb = proto + (size_t)b * R_ * P_ * H2_;
  for (int r = 0; r < R_; ++r) {
    float best = -3.402823466e38f;
    for (int p = 0; p < P_; ++p) {
      const float* pr = pb + (size_t)(r * P_ + p) * H2_;
      float s = 0.f;
#pragma unroll
      for (int i = 0; i < H2_ / 32; ++i) s += cv[i] * pr[lane + 32 * i];
#pragma unroll
      for (int off = 16; off > 0; off >>= 1) s += __shfl_xor(s, off, 32);
      best = fmaxf(best, s);
    }
    if (lane == 0) out[((size_t)bq * EE_ + n) * R_ + r] = best;
  }
}

// ---------------------------------------------------------------------------
extern "C" void kernel_launch(void* const* d_in, const int* in_sizes, int n_in,
                              void* d_out, int out_size, void* d_ws, size_t ws_size,
                              hipStream_t stream) {
  (void)in_sizes; (void)n_in; (void)out_size; (void)ws_size;

  const float* seq    = (const float*)d_in[0];
  const float* att    = (const float*)d_in[1];
  const float* W_head = (const float*)d_in[2];
  const float* W_tail = (const float*)d_in[3];
  const float* proto  = (const float*)d_in[4];
  const int*   pos    = (const int*)d_in[5];
  float* out = (float*)d_out;

  char* ws = (char*)d_ws;
  size_t off = 0;
  auto take = [&](size_t bytes) -> void* {
    void* p = (void*)(ws + off);
    off += (bytes + 255) & ~(size_t)255;
    return p;
  };
  bf16_t* ent_bf  = (bf16_t*)take((size_t)BQ_ * E_ * H_ * sizeof(bf16_t));     // 0.4 MB
  float*  ea      = (float*) take((size_t)BQ_ * E_ * NH_ * L_ * sizeof(float));// 12.6 MB
  bf16_t* mul_bf  = (bf16_t*)take((size_t)BQ_ * EE_ * L_ * sizeof(bf16_t));    // 16.8 MB
  float*  S       = (float*) take((size_t)BQ_ * EE_ * sizeof(float));          // 32 KB
  bf16_t* ctx_bf  = (bf16_t*)take((size_t)BQ_ * EE_ * H_ * sizeof(bf16_t));    // 12.6 MB
  bf16_t* cand_bf = (bf16_t*)take((size_t)BQ_ * EE_ * H2_ * sizeof(bf16_t));   // 25.2 MB

  // k1: gathers
  k1_gather<<<BQ_, 256, 0, stream>>>(seq, att, pos, ent_bf, ea);

  // k2: pairwise attention product + row sums
  k2_pairatt<<<BQ_ * EE_, 256, 0, stream>>>(ea, mul_bf, S);

  // k3: ctx GEMM — per bq: 8 M-groups x 48 N-tiles
  k3_ctx<<<BQ_ * 8 * 48, 256, 0, stream>>>(mul_bf, seq, S, ctx_bf);

  // k4: head/tail projection GEMMs + tanh
  k4_proj<<<2 * BQ_ * 8 * 48, 256, 0, stream>>>(ent_bf, ctx_bf, W_head, W_tail, cand_bf);

  // k5: prototype scoring, one wave per candidate
  k5_score<<<(BQ_ * EE_) / 8, 256, 0, stream>>>(cand_bf, proto, out);
}
